// Encoder_36507222016171
// MI455X (gfx1250) — matrix operand order repaired and checked by reading, not compiled, not run on hardware
//
#include <hip/hip_runtime.h>

// LSTM (packed-sequence) + final-hidden LayerNorm for MI455X (gfx1250).
//
//  * Fused recurrent GEMM: gates = [x_t | h] @ Wpack^T, Wpack = [W_ih ; W_hh]
//    g-major (1024 x 384) bf16, L2-resident (768 KB << 192 MB L2).
//  * 32 WGs x 16 batch rows (WMMA M=16); each WG runs all T steps privately.
//  * 8 waves/WG; wave w owns gate columns [w*128, w*128+128) = 8 N-tiles.
//  * v_wmma_f32_16x16x32_bf16, f32 accumulation; f32 cell/hidden state.
//  * Bias pre-folded into WMMA accumulator init (per-wave registers).
//  * x(t+1) warmed into cache via global_prefetch_b8 (no LOADcnt interference,
//    hides HBM latency across a full timestep).

#define B_SZ 512
#define T_SZ 512
#define I_SZ 128
#define H_SZ 256
#define G_SZ 1024   // 4*H
#define K_SZ 384    // I + H
#define A_STRIDE 392   // 16B-aligned row stride (ushorts), bank-skewed
#define G_STRIDE 1028  // gates row stride (floats), bank-skewed

typedef __attribute__((ext_vector_type(16))) __bf16          v16bf;
typedef __attribute__((ext_vector_type(8)))  float           v8f;
typedef __attribute__((ext_vector_type(8)))  unsigned short  ushortx8;
typedef __attribute__((ext_vector_type(16))) unsigned short  ushortx16;

__device__ __forceinline__ unsigned short f32_to_bf16_bits(float f) {
    unsigned int u = __builtin_bit_cast(unsigned int, f);
    u += 0x7FFFu + ((u >> 16) & 1u);          // round-to-nearest-even
    return (unsigned short)(u >> 16);
}

__device__ __forceinline__ float fast_sigmoid(float x) {
    return 1.0f / (1.0f + __expf(-x));
}
__device__ __forceinline__ float fast_tanh(float x) {
    return 2.0f / (1.0f + __expf(-2.0f * x)) - 1.0f;
}

// ---------------------------------------------------------------------------
// Prep: pack [W_ih ; W_hh] -> bf16 Wpack (g-major, 1024 x 384), bias = b_ih+b_hh
// ---------------------------------------------------------------------------
__global__ void prep_kernel(const float* __restrict__ W_ih,
                            const float* __restrict__ W_hh,
                            const float* __restrict__ b_ih,
                            const float* __restrict__ b_hh,
                            unsigned short* __restrict__ Wpack,
                            float* __restrict__ bias) {
    int idx = blockIdx.x * 256 + threadIdx.x;
    const int total = G_SZ * K_SZ;
    if (idx < total) {
        int g = idx / K_SZ;
        int k = idx - g * K_SZ;
        float w = (k < I_SZ) ? W_ih[g * I_SZ + k] : W_hh[g * H_SZ + (k - I_SZ)];
        Wpack[idx] = f32_to_bf16_bits(w);
    }
    if (idx < G_SZ) bias[idx] = b_ih[idx] + b_hh[idx];
}

// ---------------------------------------------------------------------------
// Main persistent-per-batch-tile LSTM + LayerNorm kernel
// ---------------------------------------------------------------------------
__global__ __launch_bounds__(256)
void lstm_ln_kernel(const float* __restrict__ X,
                    const int*   __restrict__ lengths,
                    const unsigned short* __restrict__ Wpack,
                    const float* __restrict__ bias,
                    const float* __restrict__ gamma,
                    const float* __restrict__ beta,
                    float* __restrict__ out) {
    __shared__ __align__(16) unsigned short A_lds[16][A_STRIDE]; // [x_t | h] bf16 bits
    __shared__ float gates_lds[16][G_STRIDE];
    __shared__ float h_lds[16][H_SZ];
    __shared__ float c_lds[16][H_SZ];
    __shared__ int   len_s[16];
    __shared__ int   maxlen_s;
    __shared__ float red1[16][16];
    __shared__ float red2[16][16];

    const int tid  = threadIdx.x;
    const int row0 = blockIdx.x * 16;

    // ---- init state, lengths ----
    for (int i = tid; i < 16 * H_SZ; i += 256) {
        int m = i >> 8, hh = i & (H_SZ - 1);
        h_lds[m][hh] = 0.0f;
        c_lds[m][hh] = 0.0f;
        A_lds[m][I_SZ + hh] = 0;   // bf16(0)
    }
    if (tid < 16) len_s[tid] = lengths[row0 + tid];
    __syncthreads();
    if (tid == 0) {
        int ml = 0;
        for (int m = 0; m < 16; ++m) ml = (len_s[m] > ml) ? len_s[m] : ml;
        maxlen_s = ml;
    }
    __syncthreads();
    const int maxlen = maxlen_s;

    const int lane  = tid & 31;
    const int wave  = tid >> 5;          // 0..7
    const int nn    = lane & 15;         // column within N-tile / row M for A-frag
    const int halfk = lane >> 4;         // K half-select per ISA layout
    const int nbase = wave * 128;        // this wave's gate-column base

    // ---- per-wave bias registers: this lane's 8 output columns (fixed) ----
    float biasreg[8];
    #pragma unroll
    for (int nt = 0; nt < 8; ++nt) biasreg[nt] = bias[nbase + nt * 16 + nn];

    for (int t = 0; t < maxlen; ++t) {
        // ---- warm x(t+1) into cache: global_prefetch_b8, no LOADcnt impact ----
        // 16 rows x 512 B = 64 cachelines of 128 B -> threads 0..63
        if (t + 1 < maxlen) {
            if (tid < 64) {
                int m   = tid >> 2;              // 0..15
                int off = (tid & 3) * 32;        // 0,32,64,96 floats = 128B lines
                const float* p =
                    &X[((size_t)(row0 + m) * T_SZ + (t + 1)) * I_SZ + off];
                __builtin_prefetch(p, 0, 3);
            }
        }

        // ---- stage x_t (16 x 128 f32 -> bf16 into A_lds cols [0,128)) ----
        #pragma unroll
        for (int r = 0; r < 8; ++r) {
            int idx = tid + r * 256;               // 0..2047
            int m = idx >> 7, ii = idx & 127;
            float xv = X[((size_t)(row0 + m) * T_SZ + t) * I_SZ + ii];
            A_lds[m][ii] = f32_to_bf16_bits(xv);
        }
        __syncthreads();   // x staged AND prev-step h writes visible

        // ---- gates(16x1024) = A(16x384) * Wpack^T, bf16 WMMA, f32 acc ----
        // accumulators start at bias (constant along M -> splat per column)
        v8f acc[8];
        #pragma unroll
        for (int nt = 0; nt < 8; ++nt) {
            v8f a;
            #pragma unroll
            for (int r = 0; r < 8; ++r) a[r] = biasreg[nt];
            acc[nt] = a;
        }

        #pragma unroll 4
        for (int kk = 0; kk < K_SZ / 32; ++kk) {
            const int k0 = kk * 32;
            // A fragment (16x32 bf16): lanes 0-15 K 0-7/16-23, lanes 16-31 K 8-15/24-31
            const unsigned short* pa = &A_lds[nn][k0 + halfk * 8];
            ushortx8 a0 = *(const ushortx8*)(pa);
            ushortx8 a1 = *(const ushortx8*)(pa + 16);
            ushortx16 av;
            #pragma unroll
            for (int j = 0; j < 8; ++j) { av[j] = a0[j]; av[8 + j] = a1[j]; }
            const v16bf afrag = __builtin_bit_cast(v16bf, av);

            #pragma unroll
            for (int nt = 0; nt < 8; ++nt) {
                // B fragment (32x16 bf16): lane nn = column g, lanes 0-15 K 0-7/16-23, lanes 16-31 K 8-15/24-31
                const int g = nbase + nt * 16 + nn;
                const unsigned short* pb =
                    Wpack + (size_t)g * K_SZ + k0 + halfk * 8;
                ushortx8 b0 = *(const ushortx8*)(pb);
                ushortx8 b1 = *(const ushortx8*)(pb + 16);
                ushortx16 bv;
                #pragma unroll
                for (int j = 0; j < 8; ++j) { bv[j] = b0[j]; bv[8 + j] = b1[j]; }
                const v16bf bfrag = __builtin_bit_cast(v16bf, bv);

                acc[nt] = __builtin_amdgcn_wmma_f32_16x16x32_bf16(
                    false, afrag, false, bfrag, (short)0, acc[nt], false, false);
            }
        }

        // ---- scatter C tiles to LDS per D-layout: VGPR r -> row r + 8*halfk ----
        const int mrow = halfk * 8;
        #pragma unroll
        for (int nt = 0; nt < 8; ++nt) {
            const int col = nbase + nt * 16 + nn;
            #pragma unroll
            for (int r = 0; r < 8; ++r)
                gates_lds[mrow + r][col] = acc[nt][r];
        }
        __syncthreads();   // gates complete; all WMMA reads of A done

        // ---- elementwise gate math + state update (f32), with length freeze ----
        {
            const int m  = tid >> 4;
            const int h0 = (tid & 15) * 16;
            if (t < len_s[m]) {
                #pragma unroll 4
                for (int j = 0; j < 16; ++j) {
                    const int hh = h0 + j;
                    float ig = fast_sigmoid(gates_lds[m][hh]);
                    float fg = fast_sigmoid(gates_lds[m][H_SZ + hh]);
                    float gg = fast_tanh   (gates_lds[m][2 * H_SZ + hh]);
                    float og = fast_sigmoid(gates_lds[m][3 * H_SZ + hh]);
                    float cn = fg * c_lds[m][hh] + ig * gg;
                    float hn = og * fast_tanh(cn);
                    c_lds[m][hh] = cn;
                    h_lds[m][hh] = hn;
                    A_lds[m][I_SZ + hh] = f32_to_bf16_bits(hn);  // feed next-step A
                }
            }
        }
        __syncthreads();
    }

    // ---- LayerNorm over H=256 per row: 16 threads/row, 16 elems/thread ----
    const int m   = tid >> 4;
    const int sub = tid & 15;
    float s = 0.0f, s2 = 0.0f;
    #pragma unroll
    for (int j = 0; j < 16; ++j) {
        float v = h_lds[m][sub * 16 + j];
        s += v; s2 += v * v;
    }
    red1[m][sub] = s;
    red2[m][sub] = s2;
    __syncthreads();
    float ts = 0.0f, ts2 = 0.0f;
    #pragma unroll
    for (int j = 0; j < 16; ++j) { ts += red1[m][j]; ts2 += red2[m][j]; }
    const float mu   = ts * (1.0f / H_SZ);
    const float var  = ts2 * (1.0f / H_SZ) - mu * mu;
    const float rstd = rsqrtf(var + 1e-5f);
    #pragma unroll
    for (int j = 0; j < 16; ++j) {
        const int hh = sub * 16 + j;
        out[(size_t)(row0 + m) * H_SZ + hh] =
            (h_lds[m][hh] - mu) * rstd * gamma[hh] + beta[hh];
    }
}

// ---------------------------------------------------------------------------
extern "C" void kernel_launch(void* const* d_in, const int* in_sizes, int n_in,
                              void* d_out, int out_size, void* d_ws, size_t ws_size,
                              hipStream_t stream) {
    (void)in_sizes; (void)n_in; (void)out_size; (void)ws_size;
    const float* X     = (const float*)d_in[0];
    const int*   len   = (const int*)d_in[1];
    const float* W_ih  = (const float*)d_in[2];
    const float* W_hh  = (const float*)d_in[3];
    const float* b_ih  = (const float*)d_in[4];
    const float* b_hh  = (const float*)d_in[5];
    const float* gamma = (const float*)d_in[6];
    const float* beta  = (const float*)d_in[7];
    float* out = (float*)d_out;

    unsigned short* Wpack = (unsigned short*)d_ws;                       // 768 KB
    float* bias = (float*)((char*)d_ws + (size_t)G_SZ * K_SZ * sizeof(unsigned short)); // 4 KB

    const int total = G_SZ * K_SZ;
    prep_kernel<<<(total + 255) / 256, 256, 0, stream>>>(W_ih, W_hh, b_ih, b_hh,
                                                         Wpack, bias);
    lstm_ln_kernel<<<B_SZ / 16, 256, 0, stream>>>(X, len, Wpack, bias,
                                                  gamma, beta, out);
}
